// LSTM_64467459113378
// MI455X (gfx1250) — matrix operand order repaired and checked by reading, not compiled, not run on hardware
//
#include <hip/hip_runtime.h>
#include <stdint.h>

typedef __attribute__((ext_vector_type(16))) __bf16 v16bf;
typedef __attribute__((ext_vector_type(8)))  float  v8f;
typedef __attribute__((ext_vector_type(4)))  float  v4f;
typedef __attribute__((ext_vector_type(4)))  __bf16 v4bf;
typedef __attribute__((ext_vector_type(4)))  int    v4i;

typedef __attribute__((address_space(1))) v4i as1_v4i;
typedef __attribute__((address_space(3))) v4i as3_v4i;

#define N_ROWS 16384
#define F_DIM  1024
#define B_DIM  512
#define MT     128   // rows per block
#define CT     32    // h-cols per block (per gate)
#define KS     32    // K per step
#define LDSP   40    // padded LDS row stride in bf16 elements (80 bytes)

union AFrag { v16bf v; uint4 q[2]; };

__device__ __forceinline__ float sigmoidf_(float x) {
    return 1.0f / (1.0f + __expf(-x));
}

// global(bf16, 16B) -> LDS, async, tracked by ASYNCcnt (in-order retirement)
#define GLD_ASYNC_B128(gp, lp) \
    __builtin_amdgcn_global_load_async_to_lds_b128((as1_v4i*)(gp), (as3_v4i*)(lp), 0, 0)
#define WAIT_ASYNC4() asm volatile("s_wait_asynccnt 0x4" ::: "memory")
#define WAIT_ASYNC0() asm volatile("s_wait_asynccnt 0x0" ::: "memory")

// ---------------------------------------------------------------------------
// fp32 -> bf16 bulk conversion (bandwidth-bound, packed v_cvt via convertvector)
// ---------------------------------------------------------------------------
__global__ __launch_bounds__(256)
void f32_to_bf16_kernel(const float* __restrict__ src,
                        unsigned short* __restrict__ dst, int n4)
{
    int i = blockIdx.x * blockDim.x + threadIdx.x;
    if (i < n4) {
        v4f v = ((const v4f*)src)[i];
        ((v4bf*)dst)[i] = __builtin_convertvector(v, v4bf);
    }
}

// ---------------------------------------------------------------------------
// Main GEMM+LSTM kernel: bf16 inputs, depth-2 async global->LDS pipeline, WMMA
// ---------------------------------------------------------------------------
__global__ __launch_bounds__(256)
void lstm_wmma_async_kernel(const unsigned short* __restrict__ hbf,
                            const unsigned short* __restrict__ xbf,
                            const unsigned short* __restrict__ whbf,
                            const unsigned short* __restrict__ wxbf,
                            const float* __restrict__ c_prev,
                            const float* __restrict__ bias,
                            float* __restrict__ h_out)
{
    __shared__ __align__(16) unsigned short sA[3][MT * LDSP];
    __shared__ __align__(16) unsigned short sB[3][MT * LDSP];

    const int tid  = threadIdx.x;
    const int wave = tid >> 5;
    const int lane = tid & 31;
    const int half = lane >> 4;
    const int ln   = lane & 15;

    const int rowb = blockIdx.y * MT;
    const int colb = blockIdx.x * CT;

    // ---- async staging addressing: 512 16B-chunks per tile, 2 per thread ----
    const int crow = tid >> 2;          // chunk row 0..63 (+64 for i=1)
    const int kc8  = (tid & 3) * 8;     // element offset of the 16B chunk in K
    int arows[2], growIdx[2], ldsOff[2];
#pragma unroll
    for (int i = 0; i < 2; ++i) {
        const int r = crow + i * 64;    // 0..127
        arows[i]  = rowb + r;
        ldsOff[i] = r * LDSP + kc8;
        growIdx[i] = (r >> 5) * F_DIM + colb + (r & 31);   // gate*F + col
    }

    v8f acc[4][2];
    {
        v8f z = {};
#pragma unroll
        for (int g = 0; g < 4; ++g)
#pragma unroll
            for (int c = 0; c < 2; ++c) acc[g][c] = z;
    }

    const int S1 = F_DIM / KS;          // 32 steps: h_prev · Wh
    const int S  = S1 + B_DIM / KS;     // +16 steps: behavior · Wx

    auto issue = [&](int s, int dbuf) {
        const unsigned short *Ap, *Wp; int lda, k0;
        if (s < S1) { Ap = hbf; Wp = whbf; lda = F_DIM; k0 = s * KS; }
        else        { Ap = xbf; Wp = wxbf; lda = B_DIM; k0 = (s - S1) * KS; }
#pragma unroll
        for (int i = 0; i < 2; ++i) {
            GLD_ASYNC_B128(Ap + (size_t)arows[i] * lda + k0 + kc8,
                           &sA[dbuf][ldsOff[i]]);
            GLD_ASYNC_B128(Wp + (size_t)growIdx[i] * lda + k0 + kc8,
                           &sB[dbuf][ldsOff[i]]);
        }
    };

    const int aRowOff = (wave * 16 + ln) * LDSP;

    auto compute = [&](int buf) {
        AFrag a;
        const unsigned short* pa = &sA[buf][aRowOff];
        a.q[0] = *(const uint4*)(pa + half * 8);
        a.q[1] = *(const uint4*)(pa + 16 + half * 8);
#pragma unroll
        for (int g = 0; g < 4; ++g) {
#pragma unroll
            for (int c = 0; c < 2; ++c) {
                AFrag bf;
                const unsigned short* pb =
                    &sB[buf][(g * 32 + c * 16 + ln) * LDSP + half * 8];
                bf.q[0] = *(const uint4*)(pb);
                bf.q[1] = *(const uint4*)(pb + 16);
                acc[g][c] = __builtin_amdgcn_wmma_f32_16x16x32_bf16(
                    false, a.v, false, bf.v, (short)0, acc[g][c], false, false);
            }
        }
    };

    // ---- depth-2 pipelined loop: two async stages in flight over WMMA ----
    issue(0, 0);
    issue(1, 1);
    WAIT_ASYNC4();          // stage 0 resident (in-order retirement)
    __syncthreads();
    int cur = 0;
#pragma unroll 1
    for (int s = 0; s < S; ++s) {
        int dst = cur + 2; if (dst >= 3) dst -= 3;
        const bool more = (s + 2 < S);
        if (more) issue(s + 2, dst);
        compute(cur);
        if (more) WAIT_ASYNC4();   // oldest stage (s+1) now resident
        else      WAIT_ASYNC0();   // drain tail
        __syncthreads();
        ++cur; if (cur == 3) cur = 0;
    }

    // ---- epilogue: bias + activations + cell blend ----
    const int wr0 = rowb + wave * 16;
#pragma unroll
    for (int c = 0; c < 2; ++c) {
        const int col = colb + c * 16 + ln;
        const float bi  = bias[0 * F_DIM + col];
        const float bff = bias[1 * F_DIM + col];
        const float bg  = bias[2 * F_DIM + col];
        const float bo  = bias[3 * F_DIM + col];
#pragma unroll
        for (int r = 0; r < 8; ++r) {
            const int row = wr0 + r + 8 * half;   // C/D layout: M = r + 8*half
            const float iv = sigmoidf_(acc[0][c][r] + bi);
            const float fv = sigmoidf_(acc[1][c][r] + bff);
            const float gv = tanhf(acc[2][c][r] + bg);
            const float ov = sigmoidf_(acc[3][c][r] + bo);
            const float cp = c_prev[(size_t)row * F_DIM + col];
            const float cn = fv * cp + iv * gv;
            h_out[(size_t)row * F_DIM + col] = ov * tanhf(cn);
        }
    }
}

// ---------------------------------------------------------------------------
// Fallback: fused convert-in-kernel version (used if d_ws is too small)
// ---------------------------------------------------------------------------
__global__ __launch_bounds__(256)
void lstm_wmma_fused_kernel(const float* __restrict__ behavior,
                            const float* __restrict__ h_prev,
                            const float* __restrict__ c_prev,
                            const float* __restrict__ Wh,
                            const float* __restrict__ Wx,
                            const float* __restrict__ bias,
                            float* __restrict__ h_out)
{
    __shared__ __align__(16) unsigned short sA[2][MT * LDSP];
    __shared__ __align__(16) unsigned short sB[2][MT * LDSP];

    const int tid  = threadIdx.x;
    const int wave = tid >> 5;
    const int lane = tid & 31;
    const int half = lane >> 4;
    const int ln   = lane & 15;

    const int rowb = blockIdx.y * MT;
    const int colb = blockIdx.x * CT;

    const int kq4  = (tid & 7) * 4;
    const int srow = tid >> 3;
    int wRowIdx[4];
#pragma unroll
    for (int i = 0; i < 4; ++i) {
        int w = srow + i * 32;
        wRowIdx[i] = (w >> 5) * F_DIM + colb + (w & 31);
    }

    v8f acc[4][2];
    {
        v8f z = {};
#pragma unroll
        for (int g = 0; g < 4; ++g)
#pragma unroll
            for (int c = 0; c < 2; ++c) acc[g][c] = z;
    }

    const int S1 = F_DIM / KS;
    const int S  = S1 + B_DIM / KS;

    v4f rA[4], rW[4];

    auto gload = [&](int s) {
        const float* Ap; const float* Wp; int lda, k0;
        if (s < S1) { Ap = h_prev;   Wp = Wh; lda = F_DIM; k0 = s * KS; }
        else        { Ap = behavior; Wp = Wx; lda = B_DIM; k0 = (s - S1) * KS; }
#pragma unroll
        for (int i = 0; i < 4; ++i) {
            rA[i] = *(const v4f*)(Ap + (size_t)(rowb + srow + i * 32) * lda + k0 + kq4);
            rW[i] = *(const v4f*)(Wp + (size_t)wRowIdx[i]             * lda + k0 + kq4);
        }
    };

    auto sstore = [&](int buf) {
#pragma unroll
        for (int i = 0; i < 4; ++i) {
            const int r = srow + i * 32;
            *(v4bf*)&sA[buf][r * LDSP + kq4] = __builtin_convertvector(rA[i], v4bf);
            *(v4bf*)&sB[buf][r * LDSP + kq4] = __builtin_convertvector(rW[i], v4bf);
        }
    };

    const int aRowOff = (wave * 16 + ln) * LDSP;

    auto compute = [&](int buf) {
        AFrag a;
        const unsigned short* pa = &sA[buf][aRowOff];
        a.q[0] = *(const uint4*)(pa + half * 8);
        a.q[1] = *(const uint4*)(pa + 16 + half * 8);
#pragma unroll
        for (int g = 0; g < 4; ++g) {
#pragma unroll
            for (int c = 0; c < 2; ++c) {
                AFrag bf;
                const unsigned short* pb =
                    &sB[buf][(g * 32 + c * 16 + ln) * LDSP + half * 8];
                bf.q[0] = *(const uint4*)(pb);
                bf.q[1] = *(const uint4*)(pb + 16);
                acc[g][c] = __builtin_amdgcn_wmma_f32_16x16x32_bf16(
                    false, a.v, false, bf.v, (short)0, acc[g][c], false, false);
            }
        }
    };

    gload(0);
    sstore(0);
    __syncthreads();
#pragma unroll 1
    for (int s = 0; s < S; ++s) {
        const int buf = s & 1;
        if (s + 1 < S) gload(s + 1);
        compute(buf);
        __syncthreads();
        if (s + 1 < S) {
            sstore(buf ^ 1);
            __syncthreads();
        }
    }

    const int wr0 = rowb + wave * 16;
#pragma unroll
    for (int c = 0; c < 2; ++c) {
        const int col = colb + c * 16 + ln;
        const float bi  = bias[0 * F_DIM + col];
        const float bff = bias[1 * F_DIM + col];
        const float bg  = bias[2 * F_DIM + col];
        const float bo  = bias[3 * F_DIM + col];
#pragma unroll
        for (int r = 0; r < 8; ++r) {
            const int row = wr0 + r + 8 * half;
            const float iv = sigmoidf_(acc[0][c][r] + bi);
            const float fv = sigmoidf_(acc[1][c][r] + bff);
            const float gv = tanhf(acc[2][c][r] + bg);
            const float ov = sigmoidf_(acc[3][c][r] + bo);
            const float cp = c_prev[(size_t)row * F_DIM + col];
            const float cn = fv * cp + iv * gv;
            h_out[(size_t)row * F_DIM + col] = ov * tanhf(cn);
        }
    }
}

extern "C" void kernel_launch(void* const* d_in, const int* in_sizes, int n_in,
                              void* d_out, int out_size, void* d_ws, size_t ws_size,
                              hipStream_t stream) {
    (void)in_sizes; (void)n_in; (void)out_size;
    const float* behavior = (const float*)d_in[0];
    const float* h_prev   = (const float*)d_in[1];
    const float* c_prev   = (const float*)d_in[2];
    const float* Wh       = (const float*)d_in[3];
    const float* Wx       = (const float*)d_in[4];
    const float* bias     = (const float*)d_in[5];
    float* h_out          = (float*)d_out;

    const size_t nH  = (size_t)N_ROWS * F_DIM;        // 16M
    const size_t nX  = (size_t)N_ROWS * B_DIM;        // 8M
    const size_t nWh = (size_t)4 * F_DIM * F_DIM;     // 4M
    const size_t nWx = (size_t)4 * F_DIM * B_DIM;     // 2M
    const size_t need = (nH + nX + nWh + nWx) * sizeof(unsigned short);

    dim3 grid(F_DIM / CT, N_ROWS / MT);   // 32 x 128
    dim3 block(256);

    if (ws_size >= need) {
        unsigned short* hbf  = (unsigned short*)d_ws;
        unsigned short* xbf  = hbf + nH;
        unsigned short* whbf = xbf + nX;
        unsigned short* wxbf = whbf + nWh;

        f32_to_bf16_kernel<<<(int)(nH  / 4 / 256), 256, 0, stream>>>(h_prev,   hbf,  (int)(nH  / 4));
        f32_to_bf16_kernel<<<(int)(nX  / 4 / 256), 256, 0, stream>>>(behavior, xbf,  (int)(nX  / 4));
        f32_to_bf16_kernel<<<(int)(nWh / 4 / 256), 256, 0, stream>>>(Wh,       whbf, (int)(nWh / 4));
        f32_to_bf16_kernel<<<(int)(nWx / 4 / 256), 256, 0, stream>>>(Wx,       wxbf, (int)(nWx / 4));

        lstm_wmma_async_kernel<<<grid, block, 0, stream>>>(hbf, xbf, whbf, wxbf,
                                                           c_prev, bias, h_out);
    } else {
        lstm_wmma_fused_kernel<<<grid, block, 0, stream>>>(behavior, h_prev, c_prev,
                                                           Wh, Wx, bias, h_out);
    }
}
